// SDPAWrapper_18992345383491
// MI455X (gfx1250) — matrix operand order repaired and checked by reading, not compiled, not run on hardware
//
#include <hip/hip_runtime.h>
#include <hip/hip_bf16.h>

#define S_LEN   4096
#define HEADS   8
#define DHEAD   128
#define QTILE   128   // queries per workgroup (8 waves x 16 rows)
#define KTILE   64    // keys per iteration
#define MAXCU   9     // max cu_seqlens entries kept in registers

typedef __attribute__((ext_vector_type(16))) __bf16 v16bf;
typedef __attribute__((ext_vector_type(8)))  __bf16 v8bf;
typedef __attribute__((ext_vector_type(2)))  __bf16 v2bf;
typedef __attribute__((ext_vector_type(8)))  float  v8f;

__global__ __launch_bounds__(256, 1)
void fa_block_causal_wmma(const float* __restrict__ Q,
                          const float* __restrict__ K,
                          const float* __restrict__ V,
                          const int*   __restrict__ cu,
                          int ncu,
                          float* __restrict__ Out)
{
    __shared__ __bf16 sK[DHEAD][KTILE];      // 16 KB, K tile transposed (d-major)
    __shared__ __bf16 sV[KTILE][DHEAD];      // 16 KB, V tile row-major
    __shared__ __bf16 sP[8][16][KTILE];      // 16 KB, per-wave P slab

    const int tid  = threadIdx.x;
    const int wid  = tid >> 5;
    const int lane = tid & 31;
    const int hlf  = lane >> 4;   // 16-lane half
    const int n0   = lane & 15;   // N index / A-row index

    const int head   = blockIdx.y;
    const int qblock = blockIdx.x * QTILE;

    // fold 1/sqrt(128) * log2(e) into Q so softmax runs in exp2 domain
    const float qscale = 0.08838834764831845f * 1.4426950408889634f;

    int cuv[MAXCU];
#pragma unroll
    for (int i = 0; i < MAXCU; ++i)
        cuv[i] = (i < ncu) ? cu[i] : 0x7FFFFFFF;

    // ---- Q tile for this wave: 4 A-operands (16x32 bf16), pre-scaled ----
    const int qa_row = qblock + wid * 16 + n0;
    const float* qptr = Q + ((size_t)head * S_LEN + qa_row) * DHEAD;

    v16bf Aq[4];
#pragma unroll
    for (int c = 0; c < 4; ++c) {
#pragma unroll
        for (int i = 0; i < 8; ++i) {
            Aq[c][i]     = (__bf16)(qptr[c * 32 +      hlf * 8 + i] * qscale);
            Aq[c][i + 8] = (__bf16)(qptr[c * 32 + 16 + hlf * 8 + i] * qscale);
        }
    }

    const int qlo = qblock + wid * 16;   // min query row of this wave
    int qdoc[8];
#pragma unroll
    for (int r = 0; r < 8; ++r) {
        int d = 0;
#pragma unroll
        for (int i = 1; i < MAXCU; ++i) d += (cuv[i] <= qlo + hlf * 8 + r) ? 1 : 0;
        qdoc[r] = d;
    }
    int qdoc_lo = 0, qdoc_hi = 0;
#pragma unroll
    for (int i = 1; i < MAXCU; ++i) {
        qdoc_lo += (cuv[i] <= qlo) ? 1 : 0;
        qdoc_hi += (cuv[i] <= qlo + 15) ? 1 : 0;
    }

    float mrow[8], lrow[8];
    v8f Oacc[8] = {};
#pragma unroll
    for (int r = 0; r < 8; ++r) { mrow[r] = -1e30f; lrow[r] = 0.0f; }

    // block-causal key range for the whole WG (uniform -> __syncthreads legal)
    int d0 = 0;
#pragma unroll
    for (int i = 1; i < MAXCU; ++i) d0 += (cuv[i] <= qblock) ? 1 : 0;
    const int kstart = cuv[d0] & ~(KTILE - 1);
    const int kend   = qblock + QTILE;

    for (int kb = kstart; kb < kend; kb += KTILE) {
        // ---- cooperative staging: K transposed (paired b32 stores) + V row-major ----
        {
            const int kp2 = tid >> 3;          // key pair 0..31 -> rows 2*kp2, 2*kp2+1
            const int db  = (tid & 7) * 16;    // d offset
            const float* k0 = K + ((size_t)head * S_LEN + kb + 2 * kp2) * DHEAD + db;
            const float* v0 = V + ((size_t)head * S_LEN + kb + 2 * kp2) * DHEAD + db;

            float a0[16], a1[16];
#pragma unroll
            for (int i = 0; i < 16; ++i) { a0[i] = k0[i]; a1[i] = k0[DHEAD + i]; }
            if (kb + KTILE < kend) {           // prefetch next tile into L2
                __builtin_prefetch(k0 + (size_t)KTILE * DHEAD, 0, 3);
                __builtin_prefetch(v0 + (size_t)KTILE * DHEAD, 0, 3);
            }
#pragma unroll
            for (int i = 0; i < 16; ++i) {
                v2bf t; t[0] = (__bf16)a0[i]; t[1] = (__bf16)a1[i];
                *(v2bf*)&sK[db + i][2 * kp2] = t;
            }
#pragma unroll
            for (int i = 0; i < 16; ++i) { a0[i] = v0[i]; a1[i] = v0[DHEAD + i]; }
            v8bf t0, t1, t2, t3;
#pragma unroll
            for (int i = 0; i < 8; ++i) {
                t0[i] = (__bf16)a0[i];     t1[i] = (__bf16)a0[8 + i];
                t2[i] = (__bf16)a1[i];     t3[i] = (__bf16)a1[8 + i];
            }
            v8bf* vd0 = (v8bf*)&sV[2 * kp2][db];
            v8bf* vd1 = (v8bf*)&sV[2 * kp2 + 1][db];
            vd0[0] = t0; vd0[1] = t1;
            vd1[0] = t2; vd1[1] = t3;
        }
        __syncthreads();

        // ---- S = Q*K^T : four 16x16 fp32 C-tiles, K-dim 128 -> 16 WMMA ----
        v8f St[4] = {};
#pragma unroll
        for (int c = 0; c < 4; ++c) {
            const __bf16* krow = &sK[c * 32][n0];
#pragma unroll
            for (int nt = 0; nt < 4; ++nt) {
                v16bf B;
#pragma unroll
                for (int i = 0; i < 8; ++i) {
                    B[i]     = krow[(hlf * 8 + i) * KTILE + nt * 16];
                    B[i + 8] = krow[(16 + hlf * 8 + i) * KTILE + nt * 16];
                }
                St[nt] = __builtin_amdgcn_wmma_f32_16x16x32_bf16(false, Aq[c], false, B, (short)0, St[nt], false, false);
            }
        }

        // ---- online softmax (exp2 domain) ----
        int kdoc_lo = 0, kdoc_hi = 0;
#pragma unroll
        for (int i = 1; i < MAXCU; ++i) {
            kdoc_lo += (cuv[i] <= kb) ? 1 : 0;
            kdoc_hi += (cuv[i] <= kb + KTILE - 1) ? 1 : 0;
        }
        const bool full = (kb + KTILE - 1 <= qlo) && (kdoc_lo == kdoc_hi) &&
                          (kdoc_lo == qdoc_lo) && (qdoc_lo == qdoc_hi);

        if (full) {
            // interior tile: no masking needed (wave-uniform branch)
#pragma unroll
            for (int r = 0; r < 8; ++r) {
                float mx = fmaxf(fmaxf(St[0][r], St[1][r]), fmaxf(St[2][r], St[3][r]));
#pragma unroll
                for (int off = 8; off >= 1; off >>= 1)
                    mx = fmaxf(mx, __shfl_xor(mx, off, 16));
                const float mnew  = fmaxf(mrow[r], mx);
                const float alpha = exp2f(mrow[r] - mnew);
                const float e0 = exp2f(St[0][r] - mnew);
                const float e1 = exp2f(St[1][r] - mnew);
                const float e2 = exp2f(St[2][r] - mnew);
                const float e3 = exp2f(St[3][r] - mnew);
                float rs = (e0 + e1) + (e2 + e3);
#pragma unroll
                for (int off = 8; off >= 1; off >>= 1)
                    rs += __shfl_xor(rs, off, 16);
                lrow[r] = lrow[r] * alpha + rs;
                mrow[r] = mnew;
#pragma unroll
                for (int j = 0; j < 8; ++j) Oacc[j][r] *= alpha;
                sP[wid][hlf * 8 + r][n0]      = (__bf16)e0;
                sP[wid][hlf * 8 + r][16 + n0] = (__bf16)e1;
                sP[wid][hlf * 8 + r][32 + n0] = (__bf16)e2;
                sP[wid][hlf * 8 + r][48 + n0] = (__bf16)e3;
            }
        } else {
            // boundary tile: causal + document masking
            int kpos[4], kdoc[4];
#pragma unroll
            for (int nt = 0; nt < 4; ++nt) {
                kpos[nt] = kb + nt * 16 + n0;
                int d = 0;
#pragma unroll
                for (int i = 1; i < MAXCU; ++i) d += (cuv[i] <= kpos[nt]) ? 1 : 0;
                kdoc[nt] = d;
            }
#pragma unroll
            for (int r = 0; r < 8; ++r) {
                const int qp = qlo + hlf * 8 + r;
                bool ok[4];
                float sv[4];
#pragma unroll
                for (int nt = 0; nt < 4; ++nt) {
                    ok[nt] = (kpos[nt] <= qp) && (kdoc[nt] == qdoc[r]);
                    sv[nt] = ok[nt] ? St[nt][r] : -1e30f;
                }
                float mx = fmaxf(fmaxf(sv[0], sv[1]), fmaxf(sv[2], sv[3]));
#pragma unroll
                for (int off = 8; off >= 1; off >>= 1)
                    mx = fmaxf(mx, __shfl_xor(mx, off, 16));
                const float mnew  = fmaxf(mrow[r], mx);
                const float alpha = exp2f(mrow[r] - mnew);
                float e[4], rs = 0.0f;
#pragma unroll
                for (int nt = 0; nt < 4; ++nt) {
                    e[nt] = ok[nt] ? exp2f(sv[nt] - mnew) : 0.0f;
                    rs += e[nt];
                }
#pragma unroll
                for (int off = 8; off >= 1; off >>= 1)
                    rs += __shfl_xor(rs, off, 16);
                lrow[r] = lrow[r] * alpha + rs;
                mrow[r] = mnew;
#pragma unroll
                for (int j = 0; j < 8; ++j) Oacc[j][r] *= alpha;
                sP[wid][hlf * 8 + r][n0]      = (__bf16)e[0];
                sP[wid][hlf * 8 + r][16 + n0] = (__bf16)e[1];
                sP[wid][hlf * 8 + r][32 + n0] = (__bf16)e[2];
                sP[wid][hlf * 8 + r][48 + n0] = (__bf16)e[3];
            }
        }
        asm volatile("s_wait_dscnt 0" ::: "memory");

        // ---- P: C layout -> two A operands (16x32 each) ----
        v16bf Ap0, Ap1;
        {
            const __bf16* prow = &sP[wid][n0][0];
            union { v16bf v; v8bf h[2]; } u;
            u.h[0] = *(const v8bf*)(prow +      hlf * 8);
            u.h[1] = *(const v8bf*)(prow + 16 + hlf * 8);
            Ap0 = u.v;
            u.h[0] = *(const v8bf*)(prow + 32 + hlf * 8);
            u.h[1] = *(const v8bf*)(prow + 48 + hlf * 8);
            Ap1 = u.v;
        }

        // ---- O += P * V : 16 WMMA across D=128 ----
#pragma unroll
        for (int j = 0; j < 8; ++j) {
            v16bf Bv0, Bv1;
#pragma unroll
            for (int i = 0; i < 8; ++i) {
                Bv0[i]     = sV[hlf * 8 + i][j * 16 + n0];
                Bv0[i + 8] = sV[16 + hlf * 8 + i][j * 16 + n0];
                Bv1[i]     = sV[32 + hlf * 8 + i][j * 16 + n0];
                Bv1[i + 8] = sV[48 + hlf * 8 + i][j * 16 + n0];
            }
            Oacc[j] = __builtin_amdgcn_wmma_f32_16x16x32_bf16(false, Ap0, false, Bv0, (short)0, Oacc[j], false, false);
            Oacc[j] = __builtin_amdgcn_wmma_f32_16x16x32_bf16(false, Ap1, false, Bv1, (short)0, Oacc[j], false, false);
        }
        __syncthreads();
    }

    // ---- epilogue: O / l ----
#pragma unroll
    for (int r = 0; r < 8; ++r) {
        const float invl = 1.0f / lrow[r];
        float* orow = Out + ((size_t)head * S_LEN + qlo + hlf * 8 + r) * DHEAD;
#pragma unroll
        for (int j = 0; j < 8; ++j)
            orow[j * 16 + n0] = Oacc[j][r] * invl;
    }
}

extern "C" void kernel_launch(void* const* d_in, const int* in_sizes, int n_in,
                              void* d_out, int out_size, void* d_ws, size_t ws_size,
                              hipStream_t stream) {
    (void)n_in; (void)out_size; (void)d_ws; (void)ws_size;
    const float* Q  = (const float*)d_in[0];
    const float* K  = (const float*)d_in[1];
    const float* V  = (const float*)d_in[2];
    const int*   cu = (const int*)d_in[3];
    const int    ncu = in_sizes[3];
    float* Out = (float*)d_out;

    dim3 grid(S_LEN / QTILE, HEADS);
    dim3 block(256);
    fa_block_causal_wmma<<<grid, block, 0, stream>>>(Q, K, V, cu, ncu, Out);
}
